// PeepholeLSTMCell_39960375722061
// MI455X (gfx1250) — matrix operand order repaired and checked by reading, not compiled, not run on hardware
//
#include <hip/hip_runtime.h>

#define UNITS  1024
#define IN_DIM 1024
#define BATCH  8192
#define KTOT   (IN_DIM + UNITS)   // 2048 (concatenated [X|H] GEMM)
#define NKSTEP (KTOT / 32)        // 64 K-steps

#define BM 128       // batch rows per block
#define BN 32        // units per block (all 4 gates computed for these units)
#define BK 32        // K per step (one bf16 WMMA K)
#define ASTR 40      // padded LDS row stride in bf16 elems (32 data + 8 pad)
#define NTHREADS 256

typedef __attribute__((ext_vector_type(16))) __bf16         v16bf;
typedef __attribute__((ext_vector_type(8)))  float          v8f;
typedef __attribute__((ext_vector_type(8)))  unsigned short u16x8;
typedef __attribute__((ext_vector_type(4)))  unsigned short u16x4;

// Split fp32 into bf16 hi (truncation) + bf16 lo (residual). hi*hi + lo*hi + hi*lo
// recovers ~16 mantissa bits relative accuracy with fp32 accumulation.
__device__ __forceinline__ void split_bf16(float x, unsigned short& h, unsigned short& l) {
    unsigned u  = __float_as_uint(x);
    unsigned hb = u & 0xFFFF0000u;
    float r = x - __uint_as_float(hb);
    h = (unsigned short)(hb >> 16);
    l = (unsigned short)(__float_as_uint(r) >> 16);
}

__device__ __forceinline__ float fsigmoid(float x) { return 1.0f / (1.0f + __expf(-x)); }
__device__ __forceinline__ float ftanh(float x) {
    float ax = fabsf(x);
    float e  = __expf(-2.0f * ax);
    float t  = (1.0f - e) / (1.0f + e);
    return copysignf(t, x);
}

union FragCast { struct { u16x8 a, b; } p; v16bf v; };

// Load a 16-element bf16 fragment as two 16B LDS reads (ds_load_b128 x2).
__device__ __forceinline__ v16bf ld_frag(const unsigned short* p, int off0, int off1) {
    FragCast u;
    u.p.a = *(const u16x8*)(p + off0);
    u.p.b = *(const u16x8*)(p + off1);
    return u.v;
}

// Resolve K-step -> (source matrices, local k offset) for the concatenated GEMM.
__device__ __forceinline__ void src_for(int ks,
                                        const float* inputs, const float* h_prev,
                                        const float* kernel_, const float* rkernel,
                                        const float*& Asrc, const float*& Bsrc, int& kk) {
    const int kg = ks * BK;
    if (kg < IN_DIM) { Asrc = inputs; Bsrc = kernel_;  kk = kg; }
    else             { Asrc = h_prev; Bsrc = rkernel;  kk = kg - IN_DIM; }
}

__global__ __launch_bounds__(NTHREADS) void
peephole_lstm_wmma(const float* __restrict__ inputs, const float* __restrict__ c_prev,
                   const float* __restrict__ h_prev, const float* __restrict__ kernel_,
                   const float* __restrict__ rkernel, const float* __restrict__ bias,
                   const float* __restrict__ pf, const float* __restrict__ pi,
                   const float* __restrict__ po,
                   float* __restrict__ out_h, float* __restrict__ out_c)
{
    // Double-buffered staging (ping-pong): 2 x ~40KB << 320KB WGP LDS pool.
    __shared__ unsigned short sAhi[2][BM * ASTR];      // A tile hi, [M][K] row-major
    __shared__ unsigned short sAlo[2][BM * ASTR];      // A tile lo
    __shared__ unsigned short sBhi[2][4 * BN * ASTR];  // B tiles hi, [gate][N][K] (transposed)
    __shared__ unsigned short sBlo[2][4 * BN * ASTR];  // B tiles lo

    const int tid  = threadIdx.x;
    const int lane = tid & 31;
    const int w    = tid >> 5;       // wave id 0..7
    const int wm   = w & 3;          // M quadrant (32 rows each)
    const int wn   = w >> 2;         // 0..1 N half (16 units each)
    const int half = lane >> 4;      // half-wave
    const int lr   = lane & 15;

    const int m0 = blockIdx.y * BM;
    const int n0 = blockIdx.x * BN;

    v8f acc[4][2];
    {
        v8f z = {};
        #pragma unroll
        for (int g = 0; g < 4; ++g)
            #pragma unroll
            for (int m = 0; m < 2; ++m)
                acc[g][m] = z;
    }

    float4 aR[4], bR[4];

    // ---- register-stage global tile for K-step `ks`
    auto load_tile = [&](int ks) {
        const float* Asrc; const float* Bsrc; int kk;
        src_for(ks, inputs, h_prev, kernel_, rkernel, Asrc, Bsrc, kk);
        #pragma unroll
        for (int it = 0; it < 4; ++it) {
            int idx = tid + it * NTHREADS;      // 0..1023 float4 slots
            int row = idx >> 3;                 // 0..127
            int c4  = (idx & 7) << 2;           // k offset 0,4,..28
            aR[it] = *(const float4*)(Asrc + (size_t)(m0 + row) * IN_DIM + kk + c4);
            int g   = idx >> 8;                 // gate 0..3
            int rem = idx & 255;
            int kr  = rem >> 3;                 // K row 0..31
            int b4  = (rem & 7) << 2;           // N col 0,4,..28
            bR[it] = *(const float4*)(Bsrc + (size_t)(kk + kr) * (4 * UNITS)
                                      + g * UNITS + n0 + b4);
        }
    };

    // ---- split + store staged registers into LDS buffer `bsel`
    auto store_tile = [&](int bsel) {
        #pragma unroll
        for (int it = 0; it < 4; ++it) {
            int idx = tid + it * NTHREADS;
            int row = idx >> 3;
            int c4  = (idx & 7) << 2;
            unsigned short h0,h1,h2,h3,l0,l1,l2,l3;
            split_bf16(aR[it].x, h0, l0); split_bf16(aR[it].y, h1, l1);
            split_bf16(aR[it].z, h2, l2); split_bf16(aR[it].w, h3, l3);
            u16x4 hv = {h0, h1, h2, h3};
            u16x4 lv = {l0, l1, l2, l3};
            *(u16x4*)&sAhi[bsel][row * ASTR + c4] = hv;
            *(u16x4*)&sAlo[bsel][row * ASTR + c4] = lv;

            int g   = idx >> 8;
            int rem = idx & 255;
            int kr  = rem >> 3;
            int b4  = (rem & 7) << 2;
            split_bf16(bR[it].x, h0, l0); split_bf16(bR[it].y, h1, l1);
            split_bf16(bR[it].z, h2, l2); split_bf16(bR[it].w, h3, l3);
            unsigned base = (unsigned)(g * BN + b4) * ASTR + kr;
            sBhi[bsel][base]            = h0; sBhi[bsel][base + ASTR]     = h1;
            sBhi[bsel][base + 2 * ASTR] = h2; sBhi[bsel][base + 3 * ASTR] = h3;
            sBlo[bsel][base]            = l0; sBlo[bsel][base + ASTR]     = l1;
            sBlo[bsel][base + 2 * ASTR] = l2; sBlo[bsel][base + 3 * ASTR] = l3;
        }
    };

    // ---- prologue: stage tile 0 into buffer 0
    load_tile(0);
    store_tile(0);
    __syncthreads();

    for (int ks = 0; ks < NKSTEP; ++ks) {
        const int cur = ks & 1;

        // Issue next tile's global loads before the WMMA burst (latency hiding),
        // and prefetch the k+2 tile toward L2 (global_prefetch_b8).
        if (ks + 1 < NKSTEP) {
            load_tile(ks + 1);
            if (ks + 2 < NKSTEP) {
                const float* Asrc; const float* Bsrc; int kk;
                src_for(ks + 2, inputs, h_prev, kernel_, rkernel, Asrc, Bsrc, kk);
                int idx = tid;
                int row = idx >> 3, c4 = (idx & 7) << 2;
                __builtin_prefetch(Asrc + (size_t)(m0 + row) * IN_DIM + kk + c4, 0, 3);
                int g = idx >> 8, rem = idx & 255, kr = rem >> 3, b4 = (rem & 7) << 2;
                __builtin_prefetch(Bsrc + (size_t)(kk + kr) * (4 * UNITS)
                                   + g * UNITS + n0 + b4, 0, 3);
            }
        }

        // ---- fragments + WMMA (bf16x3: hi*hi + lo*hi + hi*lo) from buffer `cur`
        v16bf ah[2], al[2];
        #pragma unroll
        for (int m = 0; m < 2; ++m) {
            const unsigned short* ap = &sAhi[cur][(unsigned)(wm * 32 + m * 16 + lr) * ASTR];
            const unsigned short* lp = &sAlo[cur][(unsigned)(wm * 32 + m * 16 + lr) * ASTR];
            // A frag per ISA layout: elems 0..7 -> K=8*half.., elems 8..15 -> K=16+8*half..
            ah[m] = ld_frag(ap, 8 * half, 16 + 8 * half);
            al[m] = ld_frag(lp, 8 * half, 16 + 8 * half);
        }
        #pragma unroll
        for (int g = 0; g < 4; ++g) {
            const unsigned short* bp  = &sBhi[cur][(unsigned)(g * BN + wn * 16 + lr) * ASTR];
            const unsigned short* blp = &sBlo[cur][(unsigned)(g * BN + wn * 16 + lr) * ASTR];
            // B frag per ISA layout: N=lane%16, elems 0..7 -> K=8*half.., elems 8..15 -> K=16+8*half..
            v16bf bh = ld_frag(bp,  8 * half, 16 + 8 * half);
            v16bf bl = ld_frag(blp, 8 * half, 16 + 8 * half);
            #pragma unroll
            for (int m = 0; m < 2; ++m) {
                acc[g][m] = __builtin_amdgcn_wmma_f32_16x16x32_bf16(
                    false, ah[m], false, bh, (short)0, acc[g][m], false, false);
                acc[g][m] = __builtin_amdgcn_wmma_f32_16x16x32_bf16(
                    false, al[m], false, bh, (short)0, acc[g][m], false, false);
                acc[g][m] = __builtin_amdgcn_wmma_f32_16x16x32_bf16(
                    false, ah[m], false, bl, (short)0, acc[g][m], false, false);
            }
        }

        // ---- ping-pong: store next tile into the other buffer; one barrier/iter.
        if (ks + 1 < NKSTEP) store_tile(cur ^ 1);
        __syncthreads();
    }

    // ---- fused peephole-LSTM epilogue (per-lane unit u is fixed)
    const int u = n0 + wn * 16 + lr;
    const float bf_ = bias[0 * UNITS + u];
    const float bi_ = bias[1 * UNITS + u];
    const float bc_ = bias[2 * UNITS + u];
    const float bo_ = bias[3 * UNITS + u];
    const float pfv = pf[u], piv = pi[u], pov = po[u];

    #pragma unroll
    for (int m = 0; m < 2; ++m) {
        #pragma unroll
        for (int r = 0; r < 8; ++r) {
            // C/D layout: vgpr r holds M = r (lanes 0-15) / r+8 (lanes 16-31)
            int row = m0 + wm * 32 + m * 16 + half * 8 + r;
            float cp = c_prev[(size_t)row * UNITS + u];
            float zf = acc[0][m][r] + bf_;
            float zi = acc[1][m][r] + bi_;
            float zc = acc[2][m][r] + bc_;
            float zo = acc[3][m][r] + bo_;
            float fg = fsigmoid(zf + pfv * cp);
            float ig = fsigmoid(zi + piv * cp);
            float ct = ftanh(zc);
            float c  = fg * cp + ig * ct;
            float og = fsigmoid(zo + pov * c);
            float h  = og * ftanh(c);
            out_h[(size_t)row * UNITS + u] = h;
            out_c[(size_t)row * UNITS + u] = c;
        }
    }
}

extern "C" void kernel_launch(void* const* d_in, const int* in_sizes, int n_in,
                              void* d_out, int out_size, void* d_ws, size_t ws_size,
                              hipStream_t stream) {
    (void)in_sizes; (void)n_in; (void)out_size; (void)d_ws; (void)ws_size;
    const float* inputs = (const float*)d_in[0];
    const float* c_prev = (const float*)d_in[1];
    const float* h_prev = (const float*)d_in[2];
    const float* kern   = (const float*)d_in[3];
    const float* rkern  = (const float*)d_in[4];
    const float* bias   = (const float*)d_in[5];
    const float* pf     = (const float*)d_in[6];
    const float* pi     = (const float*)d_in[7];
    const float* po     = (const float*)d_in[8];

    float* out_h = (float*)d_out;
    float* out_c = out_h + (size_t)BATCH * UNITS;

    dim3 grid(UNITS / BN, BATCH / BM);   // 32 x 64 = 2048 blocks
    hipLaunchKernelGGL(peephole_lstm_wmma, grid, dim3(NTHREADS), 0, stream,
                       inputs, c_prev, h_prev, kern, rkern, bias, pf, pi, po,
                       out_h, out_c);
}
